// Node2EdgeLayer_41910290874864
// MI455X (gfx1250) — matrix operand order repaired and checked by reading, not compiled, not run on hardware
//
#include <hip/hip_runtime.h>

typedef __attribute__((ext_vector_type(16))) __bf16 v16bf;
typedef __attribute__((ext_vector_type(8)))  float  v8f;

#define JNT  24
#define CDIM 64

__device__ __forceinline__ __bf16 bf(float f) { return (__bf16)f; }

__device__ __forceinline__ v8f wmma_bf16(v16bf a, v16bf b, v8f c) {
  // D = A(16x32) * B(32x16) + C, f32 accumulate
  return __builtin_amdgcn_wmma_f32_16x16x32_bf16(false, a, false, b, (short)0, c, false, false);
}

union AFrag { v16bf v; __bf16 h[16]; };
union Pack8 { uint4 q; __bf16 h[8]; };

__global__ __launch_bounds__(128)
void n2e_wmma_kernel(const float* __restrict__ node, const float* __restrict__ edge,
                     const float* __restrict__ adj,
                     const float* __restrict__ Wp, const float* __restrict__ bp,
                     const float* __restrict__ Wr, const float* __restrict__ br,
                     const float* __restrict__ Wc, const float* __restrict__ bc,
                     float* __restrict__ out, int B)
{
  // Pre-swizzled weight B-fragments (scaled by -log2(e)/3): [proj][kt][nt][lane]
  __shared__ v16bf sW[3][2][4][32];                 // 24 KB
  // Per-wave P staging, column-major: [wave][col d (64)][row k (32)] bf16
  __shared__ alignas(64) __bf16 sP[4][64][32];      // 16 KB

  const int tid  = threadIdx.x;
  const int wave = tid >> 5;
  const int lane = tid & 31;
  const int n    = lane & 15;   // tile row (A) / tile col (B,C,D)
  const int hi   = lane >> 4;

  const float kSig = -0.48089834696298783f;   // -log2(e)/3 : folds /3 and sigmoid's exp2 scale

  // ---------------- cooperative weight fragment fill (scaled) ----------------
  // B-frag layout (32x16, K x N): lane n=l%16, hi=l/16; element e -> K = 16*hi + e
  {
    const float* Wm[3] = {Wp, Wr, Wc};
    for (int it = 0; it < 6; ++it) {
      int t    = tid + it * 128;      // 768 lane-fragments total
      int fl   = t & 31;
      int rest = t >> 5;
      int fnt  = rest & 3;
      int fkt  = (rest >> 2) & 1;
      int fp   = rest >> 3;
      int fn   = fl & 15, fh = fl >> 4;
      const float* W = Wm[fp];
      AFrag fr;
      #pragma unroll
      for (int e = 0; e < 16; ++e)
        fr.h[e] = bf(W[(32 * fkt + 16 * fh + e) * CDIM + 16 * fnt + fn] * kSig);
      sW[fp][fkt][fnt][fl] = fr.v;
    }
  }

  // ---------------- per-wave constant P rows 24..31 (written once) ----------------
  // row 24 = kSig*b_parent (multiplied by adj column-sum via aT col 24)
  // row 25 = kSig*(b_rec + b_child) (multiplied by 1.0 via aT col 25); rows 26..31 = 0
  #pragma unroll
  for (int cc = 0; cc < 2; ++cc) {
    int c = lane + 32 * cc;
    Pack8 pk;
    pk.h[0] = bf(bp[c] * kSig);
    pk.h[1] = bf((br[c] + bc[c]) * kSig);
    #pragma unroll
    for (int r = 2; r < 8; ++r) pk.h[r] = (__bf16)0.0f;
    *(uint4*)&sP[wave][c][24] = pk.q;
  }
  __syncthreads();

  const int bstride = gridDim.x * 4;
  for (int b = blockIdx.x * 4 + wave; b < B; b += bstride) {
    const float* nodeB = node + (size_t)b * JNT * CDIM;
    const float* edgeB = edge + (size_t)b * JNT * CDIM;
    const float* adjB  = adj  + (size_t)b * JNT * JNT;

    // prefetch next iteration's tiles (global_prefetch_b8, spread across lanes)
    {
      int bn = b + bstride;
      if (bn < B) {
        const float* pn = node + (size_t)bn * JNT * CDIM;
        const float* pe = edge + (size_t)bn * JNT * CDIM;
        const float* pa_ = adj + (size_t)bn * JNT * JNT;
        __builtin_prefetch(pn + lane * 48, 0, 1);   // 24*64*4B = 6 KB
        __builtin_prefetch(pe + lane * 48, 0, 1);
        __builtin_prefetch(pa_ + lane * 18, 0, 1);  // 24*24*4B = 2.25 KB
      }
    }

    // ---------------- A fragments: node & edge (f32 -> bf16) ----------------
    // A layout (16x32): lane m=l%16, hi=l/16; element e -> K = 16*hi + e
    v16bf aN[2][2], aE[2][2];
    #pragma unroll
    for (int mt = 0; mt < 2; ++mt) {
      int j = 16 * mt + n;
      bool valid = (j < JNT);
      const float* nr = nodeB + j * CDIM;
      const float* er = edgeB + j * CDIM;
      #pragma unroll
      for (int kt = 0; kt < 2; ++kt) {
        AFrag fnv, fev;
        #pragma unroll
        for (int g = 0; g < 2; ++g) {
          int c0 = 32 * kt + 16 * hi + 8 * g;
          float4 x0{0.f,0.f,0.f,0.f}, x1{0.f,0.f,0.f,0.f};
          float4 y0{0.f,0.f,0.f,0.f}, y1{0.f,0.f,0.f,0.f};
          if (valid) {
            x0 = *(const float4*)(nr + c0);  x1 = *(const float4*)(nr + c0 + 4);
            y0 = *(const float4*)(er + c0);  y1 = *(const float4*)(er + c0 + 4);
          }
          fnv.h[g*8+0]=bf(x0.x); fnv.h[g*8+1]=bf(x0.y); fnv.h[g*8+2]=bf(x0.z); fnv.h[g*8+3]=bf(x0.w);
          fnv.h[g*8+4]=bf(x1.x); fnv.h[g*8+5]=bf(x1.y); fnv.h[g*8+6]=bf(x1.z); fnv.h[g*8+7]=bf(x1.w);
          fev.h[g*8+0]=bf(y0.x); fev.h[g*8+1]=bf(y0.y); fev.h[g*8+2]=bf(y0.z); fev.h[g*8+3]=bf(y0.w);
          fev.h[g*8+4]=bf(y1.x); fev.h[g*8+5]=bf(y1.y); fev.h[g*8+6]=bf(y1.z); fev.h[g*8+7]=bf(y1.w);
        }
        aN[mt][kt] = fnv.v;
        aE[mt][kt] = fev.v;
      }
    }

    // ---------------- A fragments: adj^T padded to 32x32 ----------------
    // A_bmm[j][k] = adj[b][k][j]; col 24 = colsum_j (scales b_parent), col 25 = 1 (adds b_rec+b_child)
    v16bf aT[2];
    #pragma unroll
    for (int mt = 0; mt < 2; ++mt) {
      int j = 16 * mt + n;
      bool jv = (j < JNT);
      float arr[16];
      float partial = 0.f;
      #pragma unroll
      for (int g = 0; g < 2; ++g) {
        #pragma unroll
        for (int i = 0; i < 8; ++i) {
          int k = 16 * hi + 8 * g + i;
          float v = (jv && k < JNT) ? adjB[k * JNT + j] : 0.f;
          arr[g*8+i] = v;
          partial += v;
        }
      }
      // colsum_j = sum over k=0..23 of adj[k][j]; other half-wave holds the other 8/16 terms
      float colsum = partial + __shfl_xor(partial, 16, 32);
      if (hi == 1) {                     // this half owns k=24 (e=8) and k=25 (e=9)
        arr[8] = jv ? colsum : 0.f;
        arr[9] = jv ? 1.f    : 0.f;
      }
      AFrag fr;
      #pragma unroll
      for (int e = 0; e < 16; ++e) fr.h[e] = bf(arr[e]);
      aT[mt] = fr.v;
    }

    // ---------------- P = node @ (Wp*kSig), stage to LDS column-major ----------------
    #pragma unroll
    for (int nt = 0; nt < 4; ++nt) {
      #pragma unroll
      for (int mt = 0; mt < 2; ++mt) {
        v8f p = {};
        #pragma unroll
        for (int kt = 0; kt < 2; ++kt)
          p = wmma_bf16(aN[mt][kt], sW[0][kt][nt][lane], p);
        if (!(mt == 1 && hi == 1)) {     // rows 24..31 are the constant bias rows
          Pack8 pk;
          #pragma unroll
          for (int r = 0; r < 8; ++r) pk.h[r] = bf(p[r]);
          *(uint4*)&sP[wave][16 * nt + n][16 * mt + 8 * hi] = pk.q;
        }
      }
    }
    // Same-wave DS ops are in-order (ISA ch.11); stop compiler reordering only.
    asm volatile("" ::: "memory");

    // ------- per-ntile: S = node@Wc' + edge@Wr', pa = adj^T@P' + S (all pre-scaled) -------
    float* outB = out + (size_t)b * JNT * CDIM;
    #pragma unroll
    for (int nt = 0; nt < 4; ++nt) {
      v16bf pb = *(const v16bf*)&sP[wave][16 * nt + n][16 * hi];
      #pragma unroll
      for (int mt = 0; mt < 2; ++mt) {
        v8f s = {};
        #pragma unroll
        for (int kt = 0; kt < 2; ++kt) {
          s = wmma_bf16(aN[mt][kt], sW[2][kt][nt][lane], s);   // children
          s = wmma_bf16(aE[mt][kt], sW[1][kt][nt][lane], s);   // recurrent
        }
        v8f pa = wmma_bf16(aT[mt], pb, s);   // + aggregation + all biases, pre-scaled by kSig
        // rows valid iff !(mt==1 && hi==1): uniform per half-wave -> single exec toggle
        if (!(mt == 1 && hi == 1)) {
          int j0 = 16 * mt + 8 * hi;
          float* ocol = outB + 16 * nt + n;
          #pragma unroll
          for (int r = 0; r < 8; ++r) {
            // sigmoid(sum/3) = rcp(1 + exp2(kSig*sum)), kSig folded into weights/biases
            float o = __builtin_amdgcn_rcpf(1.0f + __builtin_amdgcn_exp2f(pa[r]));
            ocol[(j0 + r) * CDIM] = o;
          }
        }
      }
    }
    asm volatile("" ::: "memory");   // keep next iteration's sP stores after this one's loads
  }
}

extern "C" void kernel_launch(void* const* d_in, const int* in_sizes, int n_in,
                              void* d_out, int out_size, void* d_ws, size_t ws_size,
                              hipStream_t stream) {
  const float* node = (const float*)d_in[0];
  const float* edge = (const float*)d_in[1];
  const float* adj  = (const float*)d_in[2];
  const float* Wp   = (const float*)d_in[3];
  const float* bp   = (const float*)d_in[4];
  const float* Wr   = (const float*)d_in[5];
  const float* br   = (const float*)d_in[6];
  const float* Wc   = (const float*)d_in[7];
  const float* bc   = (const float*)d_in[8];
  float* outp = (float*)d_out;

  int B = in_sizes[0] / (JNT * CDIM);        // 16384
  int blocks = (B + 15) / 16;                // 4 waves/WG, 4 batches per wave
  dim3 grid(blocks), block(128);
  hipLaunchKernelGGL(n2e_wmma_kernel, grid, block, 0, stream,
                     node, edge, adj, Wp, bp, Wr, br, Wc, bc, outp, B);
}
